// LSTM_80951543595052
// MI455X (gfx1250) — matrix operand order repaired and checked by reading, not compiled, not run on hardware
//
#include <hip/hip_runtime.h>
#include <hip/hip_bf16.h>

typedef __attribute__((ext_vector_type(16))) _Float16 v16h;
typedef __attribute__((ext_vector_type(8)))  _Float16 v8h;
typedef __attribute__((ext_vector_type(8)))  float    v8f;

constexpr int B = 64, S = 512, D = 1024, H = 1024, L = 2;
constexpr int G4H = 4 * H;      // 4096 gate columns (i, j, f, o)
constexpr int KW  = D + H;      // 2048: Wt row length (k dimension)
constexpr int MX  = B * S;      // 32768 rows for the x-projection GEMM
constexpr int HP  = H + 8;      // LDS row pad (16 B) to spread banks

// ---- WMMA fragment loaders (wave32, 16-bit A/B layouts per CDNA5 ISA 7.12.2) ----

// A (16x32 f16): lanes 0-15 row M=lane, K={kk..kk+7, kk+16..kk+23};
//                lanes 16-31 row M=lane-16, K={kk+8..kk+15, kk+24..kk+31}.
__device__ inline v16h load_a_frag(const _Float16* __restrict__ A, size_t lda,
                                   int m0, int kk, int lane) {
    int row   = m0 + (lane & 15);
    int khalf = (lane & 16) ? 8 : 0;
    const _Float16* p = A + (size_t)row * lda + kk + khalf;
    v8h lo = *(const v8h*)(p);
    v8h hi = *(const v8h*)(p + 16);
    return __builtin_shufflevector(lo, hi, 0,1,2,3,4,5,6,7,8,9,10,11,12,13,14,15);
}

// B (32x16 f16) from pre-transposed Wt[4H][KW]: lane = output column,
// lanes 0-15 hold K={kk..kk+7, kk+16..kk+23}, lanes 16-31 hold K={kk+8..kk+15, kk+24..kk+31}.
__device__ inline v16h load_b_frag(const _Float16* __restrict__ Wt,
                                   int n_global, int k_global, int lane) {
    int col = n_global + (lane & 15);
    int ko  = k_global + ((lane & 16) ? 8 : 0);
    const _Float16* q = Wt + (size_t)col * KW + ko;
    v8h lo = *(const v8h*)(q);
    v8h hi = *(const v8h*)(q + 16);
    return __builtin_shufflevector(lo, hi, 0,1,2,3,4,5,6,7,8,9,10,11,12,13,14,15);
}

// ---- prep kernels ----

__global__ void cast_f32_f16(const float* __restrict__ in,
                             _Float16* __restrict__ out, size_t n) {
    size_t i = (size_t)blockIdx.x * blockDim.x + threadIdx.x;
    if (i < n) out[i] = (_Float16)in[i];
}

// W: [L, KW, 4H] fp32 row-major  ->  Wt: [L, 4H, KW] f16 (N-major, contiguous K)
__global__ void transpose_w(const float* __restrict__ W, _Float16* __restrict__ Wt) {
    size_t i = (size_t)blockIdx.x * blockDim.x + threadIdx.x;
    size_t total = (size_t)L * G4H * KW;
    if (i >= total) return;
    int k = (int)(i % KW);
    size_t r = i / KW;
    int n = (int)(r % G4H);
    int l = (int)(r / G4H);
    Wt[i] = (_Float16)W[((size_t)l * KW + k) * G4H + n];
}

// ---- x-projection GEMM: out[MX,4H] = A[MX,1024](f16) @ Wt[:,0:D]^T + bias ----
__global__ void gemm_xproj(const _Float16* __restrict__ A,
                           const _Float16* __restrict__ Wt,
                           const float* __restrict__ bias,
                           float* __restrict__ out) {
    int lane = threadIdx.x & 31;
    int wid  = blockIdx.x * (blockDim.x >> 5) + (threadIdx.x >> 5);
    constexpr int NG = G4H / 64;
    int m0 = (wid / NG) * 16;
    int n0 = (wid % NG) * 64;

    v8f acc[4] = {};
    for (int kk = 0; kk < D; kk += 32) {
        v16h a = load_a_frag(A, D, m0, kk, lane);
#pragma unroll
        for (int s = 0; s < 4; ++s) {
            v16h b = load_b_frag(Wt, n0 + s * 16, kk, lane);
            acc[s] = __builtin_amdgcn_wmma_f32_16x16x32_f16(
                false, a, false, b, (short)0, acc[s], false, false);
        }
    }
    int n_lane = lane & 15;
    int m_off  = (lane & 16) ? 8 : 0;
#pragma unroll
    for (int s = 0; s < 4; ++s) {
        int n = n0 + s * 16 + n_lane;
        float bv = bias[n];
#pragma unroll
        for (int e = 0; e < 8; ++e) {
            int m = m0 + m_off + e;
            out[(size_t)m * G4H + n] = acc[s][e] + bv;
        }
    }
}

// ---- persistent per-layer recurrence ----
// grid = 4 blocks (one per 16-row batch tile), block = 1024 threads = 32 waves.
// Wave w owns H-columns [32w, 32w+32) for its batch tile across ALL timesteps:
//  - h tile (16x1024 f16) lives in LDS (A-frags via ds_load, updated via ds_store)
//  - c lives in registers for the whole sequence (c0 in once, c_fin out once)
//  - weights + precomputed xproj stream from L2/HBM
__global__ void __launch_bounds__(1024)
lstm_layer_persist(const float* __restrict__ xproj,
                   const _Float16* __restrict__ Wt,
                   const float* __restrict__ c0,
                   const float* __restrict__ h0,
                   float* __restrict__ hseq_f32, int write_f32,
                   _Float16* __restrict__ hseq_f16, int write_f16,
                   float* __restrict__ c_fin,
                   float* __restrict__ h_fin) {
    __shared__ __align__(16) _Float16 hbuf[16][HP];

    int tid  = threadIdx.x;
    int lane = tid & 31;
    int w    = tid >> 5;                 // 0..31
    int m0   = blockIdx.x * 16;          // batch-tile base
    int n0   = w * 32;                   // two 16-col subtiles

    // init h tile in LDS from h0 (fp32 -> f16)
    for (int i = tid; i < 16 * H; i += 1024) {
        int r = i >> 10, c = i & (H - 1);
        hbuf[r][c] = (_Float16)h0[(size_t)(m0 + r) * H + c];
    }

    int n_lane = lane & 15;
    int m_off  = (lane & 16) ? 8 : 0;
    int khalf  = (lane & 16) ? 8 : 0;

    float creg[2][8];
#pragma unroll
    for (int nt = 0; nt < 2; ++nt)
#pragma unroll
        for (int e = 0; e < 8; ++e)
            creg[nt][e] = c0[(size_t)(m0 + m_off + e) * H + n0 + nt * 16 + n_lane];
    __syncthreads();

    for (int t = 0; t < S; ++t) {
        v8f acc[2][4] = {};
        for (int kk = 0; kk < H; kk += 32) {
            const _Float16* p = &hbuf[n_lane][kk + khalf];     // A row = lane&15
            v8h lo = *(const v8h*)p;
            v8h hi = *(const v8h*)(p + 16);
            v16h a = __builtin_shufflevector(lo, hi,
                         0,1,2,3,4,5,6,7,8,9,10,11,12,13,14,15);
#pragma unroll
            for (int nt = 0; nt < 2; ++nt)
#pragma unroll
                for (int g = 0; g < 4; ++g) {                  // i, j, f, o
                    v16h b = load_b_frag(Wt, g * H + n0 + nt * 16, D + kk, lane);
                    acc[nt][g] = __builtin_amdgcn_wmma_f32_16x16x32_f16(
                        false, a, false, b, (short)0, acc[nt][g], false, false);
                }
        }
        __syncthreads();                 // all waves done reading hbuf for step t

#pragma unroll
        for (int nt = 0; nt < 2; ++nt) {
            int n = n0 + nt * 16 + n_lane;
#pragma unroll
            for (int e = 0; e < 8; ++e) {
                int mloc = m_off + e;
                int m = m0 + mloc;
                size_t xrow = ((size_t)m * S + t) * (size_t)G4H;
                float gi = acc[nt][0][e] + xproj[xrow + 0 * H + n];
                float gj = acc[nt][1][e] + xproj[xrow + 1 * H + n];
                float gf = acc[nt][2][e] + xproj[xrow + 2 * H + n];
                float go = acc[nt][3][e] + xproj[xrow + 3 * H + n];

                float si = 1.0f / (1.0f + expf(-gi));
                float sf = 1.0f / (1.0f + expf(-gf));
                float so = 1.0f / (1.0f + expf(-go));
                float cn = creg[nt][e] * sf + si * tanhf(gj);
                float hn = tanhf(cn) * so;
                creg[nt][e] = cn;

                hbuf[mloc][n] = (_Float16)hn;                  // next-step A
                size_t hidx = ((size_t)m * S + t) * H + n;
                if (write_f16) hseq_f16[hidx] = (_Float16)hn;  // layer-1 GEMM input
                if (write_f32) hseq_f32[hidx] = hn;            // final-layer output
                if (t == S - 1) h_fin[(size_t)m * H + n] = hn;
            }
        }
        __syncthreads();                 // hbuf updated before next step reads
    }

#pragma unroll
    for (int nt = 0; nt < 2; ++nt)
#pragma unroll
        for (int e = 0; e < 8; ++e)
            c_fin[(size_t)(m0 + m_off + e) * H + n0 + nt * 16 + n_lane] = creg[nt][e];
}

extern "C" void kernel_launch(void* const* d_in, const int* in_sizes, int n_in,
                              void* d_out, int out_size, void* d_ws, size_t ws_size,
                              hipStream_t stream) {
    (void)in_sizes; (void)n_in; (void)out_size; (void)ws_size;
    const float* x    = (const float*)d_in[0];   // [B, S, D]
    const float* c0   = (const float*)d_in[1];   // [L, B, H]
    const float* h0   = (const float*)d_in[2];   // [L, B, H]
    const float* W    = (const float*)d_in[3];   // [L, D+H, 4H]
    const float* bias = (const float*)d_in[4];   // [L, 4H]
    float* out = (float*)d_out;

    // workspace carve-up
    char* ws = (char*)d_ws;
    size_t off = 0;
    _Float16* xh     = (_Float16*)(ws + off); off += (size_t)MX * D * 2;       // 64 MB
    _Float16* hseq16 = (_Float16*)(ws + off); off += (size_t)MX * H * 2;       // 64 MB
    _Float16* Wt     = (_Float16*)(ws + off); off += (size_t)L * G4H * KW * 2; // 33 MB
    float*    xproj  = (float*)(ws + off);    // MX * 4H * 4 = 512 MB

    float* out_hseq = out;                            // [B, S, H]
    float* out_cfin = out + (size_t)B * S * H;        // [L, B, H]
    float* out_hfin = out_cfin + (size_t)L * B * H;   // [L, B, H]

    {   size_t n = (size_t)MX * D;
        cast_f32_f16<<<(unsigned)((n + 255) / 256), 256, 0, stream>>>(x, xh, n); }
    {   size_t n = (size_t)L * G4H * KW;
        transpose_w<<<(unsigned)((n + 255) / 256), 256, 0, stream>>>(W, Wt); }

    const int xproj_blocks = (MX / 16) * (G4H / 64) / 8;   // 8 waves / 256-thread block

    for (int l = 0; l < L; ++l) {
        const _Float16* Al  = (l == 0) ? xh : hseq16;      // layer-1 input = layer-0 h
        const _Float16* Wtl = Wt + (size_t)l * G4H * KW;

        gemm_xproj<<<xproj_blocks, 256, 0, stream>>>(Al, Wtl,
                                                     bias + (size_t)l * G4H, xproj);

        lstm_layer_persist<<<B / 16, 1024, 0, stream>>>(
            xproj, Wtl,
            c0 + (size_t)l * B * H,
            h0 + (size_t)l * B * H,
            out_hseq, (l == L - 1) ? 1 : 0,    // fp32 h_seq only for final layer
            hseq16,   (l == 0) ? 1 : 0,        // f16 h_seq only feeds layer 1
            out_cfin + (size_t)l * B * H,
            out_hfin + (size_t)l * B * H);
    }
}
